// MAMCA_36859409334278
// MI455X (gfx1250) — matrix operand order repaired and checked by reading, not compiled, not run on hardware
//
#include <hip/hip_runtime.h>
#include <hip/hip_bf16.h>

// ---------------------------------------------------------------------------
// Mamba forward on gfx1250 (MI455X). GEMMs via v_wmma_f32_16x16x32_f16.
// Wave tile: 32(M) x 64(N) -> 8 WMMAs per 32-wide K-step, 2x A/B reuse.
// ---------------------------------------------------------------------------

#define BATCH   16
#define SEQL    2048
#define MTOT    (BATCH * SEQL)      // 32768
#define CIN_    6
#define DMODEL  512
#define DINNER  1024
#define DSTATE  16
#define DCONV   4
#define DTRANK  32
#define NLAYER  2
#define NCLS    32

typedef __attribute__((ext_vector_type(16))) _Float16 v16h;
typedef __attribute__((ext_vector_type(8)))  _Float16 v8h;
typedef __attribute__((ext_vector_type(8)))  float    v8f;

// A fragment (16x32 f16, contiguous K to match B): lanes 0-15 carry K {kb..kb+15},
// lanes 16-31 carry K {kb+16..kb+31}.
__device__ __forceinline__ v16h load_afrag(const _Float16* __restrict__ arow, int kb, int hf)
{
    v8h lo = *(const v8h*)(arow + kb + 16 * hf);
    v8h hi = *(const v8h*)(arow + kb + 16 * hf + 8);
    v16h a;
    #pragma unroll
    for (int i = 0; i < 8; ++i) { a[i] = lo[i]; a[8 + i] = hi[i]; }
    return a;
}

// ---------------------------------------------------------------------------
// Generic WMMA GEMM: C[M,N] (f32) = A[M,K](f16) * W[N,K](f16)^T  (+ epilogue)
// Wave computes a 32(M) x 64(N) tile: 2 A fragments x 4 B fragments -> 8 WMMAs
// per K-step. Block = 4 waves stacked in M -> 128x64 tile. Grid = (M/128, N/64).
// mode 0: C = acc            mode 2: C = softplus(acc + bias[n])
// mode 3: C += acc (residual)
// ---------------------------------------------------------------------------
__global__ __launch_bounds__(128)
void mamba_gemm_wmma(const _Float16* __restrict__ A,
                     const _Float16* __restrict__ W,
                     float* __restrict__ Cout,
                     const float* __restrict__ bias,
                     int M, int N, int K, int mode)
{
    const int lane = threadIdx.x & 31;
    const int wave = threadIdx.x >> 5;
    const int m0   = (blockIdx.x * 4 + wave) * 32;
    const int n0   = blockIdx.y * 64;
    if (m0 >= M) return;

    const int hf  = lane >> 4;     // 0: lanes 0-15, 1: lanes 16-31
    const int l15 = lane & 15;

    v8f acc[2][4] = {};

    const _Float16* __restrict__ arow0 = A + (size_t)(m0 + l15) * K;
    const _Float16* __restrict__ arow1 = A + (size_t)(m0 + 16 + l15) * K;
    const _Float16* __restrict__ wbase = W + (size_t)(n0 + l15) * K;

    for (int kb = 0; kb < K; kb += 32) {
        v16h a0 = load_afrag(arow0, kb, hf);
        v16h a1 = load_afrag(arow1, kb, hf);

        // B fragment (32x16 f16): lane n holds column n, K kb..kb+15 (lanes 0-15)
        // or K kb+16..kb+31 (lanes 16-31) -> contiguous 32B from row-major W[N,K].
        const int koff = kb + 16 * hf;
        v16h b0 = *(const v16h*)(wbase + (size_t) 0 * K + koff);
        v16h b1 = *(const v16h*)(wbase + (size_t)16 * K + koff);
        v16h b2 = *(const v16h*)(wbase + (size_t)32 * K + koff);
        v16h b3 = *(const v16h*)(wbase + (size_t)48 * K + koff);

        acc[0][0] = __builtin_amdgcn_wmma_f32_16x16x32_f16(false, a0, false, b0, (short)0, acc[0][0], false, false);
        acc[0][1] = __builtin_amdgcn_wmma_f32_16x16x32_f16(false, a0, false, b1, (short)0, acc[0][1], false, false);
        acc[0][2] = __builtin_amdgcn_wmma_f32_16x16x32_f16(false, a0, false, b2, (short)0, acc[0][2], false, false);
        acc[0][3] = __builtin_amdgcn_wmma_f32_16x16x32_f16(false, a0, false, b3, (short)0, acc[0][3], false, false);
        acc[1][0] = __builtin_amdgcn_wmma_f32_16x16x32_f16(false, a1, false, b0, (short)0, acc[1][0], false, false);
        acc[1][1] = __builtin_amdgcn_wmma_f32_16x16x32_f16(false, a1, false, b1, (short)0, acc[1][1], false, false);
        acc[1][2] = __builtin_amdgcn_wmma_f32_16x16x32_f16(false, a1, false, b2, (short)0, acc[1][2], false, false);
        acc[1][3] = __builtin_amdgcn_wmma_f32_16x16x32_f16(false, a1, false, b3, (short)0, acc[1][3], false, false);
    }

    // C/D layout: lanes 0-15 -> N=lane, M = base + vgpr; lanes 16-31 -> N=lane-16, M = base+8+vgpr.
    #pragma unroll
    for (int r = 0; r < 2; ++r) {
        #pragma unroll
        for (int s = 0; s < 4; ++s) {
            const int nc = n0 + s * 16 + l15;
            #pragma unroll
            for (int i = 0; i < 8; ++i) {
                const int row = m0 + 16 * r + 8 * hf + i;
                const size_t o = (size_t)row * N + nc;
                float v = acc[r][s][i];
                if (mode == 0) {
                    Cout[o] = v;
                } else if (mode == 2) {
                    float t = v + bias[nc];
                    Cout[o] = (t > 20.f) ? t : log1pf(expf(t));   // softplus
                } else {
                    Cout[o] += v;                                  // residual add
                }
            }
        }
    }
}

// ---------------------------------------------------------------------------
// f32 -> f16 elementwise convert (weights + activations)
// ---------------------------------------------------------------------------
__global__ void cvt_f32_f16(const float* __restrict__ in, _Float16* __restrict__ out, int n)
{
    int i = blockIdx.x * blockDim.x + threadIdx.x;
    if (i < n) out[i] = (_Float16)in[i];
}

// ---------------------------------------------------------------------------
// Stem: depthwise conv3 (pad 1) over x(B,CIN,L) -> tmp
// ---------------------------------------------------------------------------
__global__ void stem_dw(const float* __restrict__ x, const float* __restrict__ dw,
                        float* __restrict__ tmp)
{
    int i = blockIdx.x * blockDim.x + threadIdx.x;
    if (i >= BATCH * CIN_ * SEQL) return;
    int l = i % SEQL;
    int c = (i / SEQL) % CIN_;
    float acc = 0.f;
    #pragma unroll
    for (int k = 0; k < 3; ++k) {
        int ls = l - 1 + k;
        if (ls >= 0 && ls < SEQL) acc += dw[c * 3 + k] * x[i + (ls - l)];
    }
    tmp[i] = acc;
}

// Pointwise conv (CIN->DMODEL) + BN + exact GELU -> h (MTOT, DMODEL) row-major
__global__ void stem_pw(const float* __restrict__ tmp, const float* __restrict__ pw,
                        const float* __restrict__ bnw, const float* __restrict__ bnb,
                        const float* __restrict__ bnm, const float* __restrict__ bnv,
                        float* __restrict__ h)
{
    int i = blockIdx.x * blockDim.x + threadIdx.x;
    if (i >= MTOT * DMODEL) return;
    int dm = i % DMODEL;
    int m  = i / DMODEL;
    int b  = m / SEQL;
    int l  = m % SEQL;
    float pre = 0.f;
    #pragma unroll
    for (int c = 0; c < CIN_; ++c)
        pre += tmp[((size_t)(b * CIN_ + c)) * SEQL + l] * pw[dm * CIN_ + c];
    float scale = bnw[dm] * rsqrtf(bnv[dm] + 1e-5f);
    float v = pre * scale + (bnb[dm] - bnm[dm] * scale);
    h[i] = 0.5f * v * (1.f + erff(v * 0.70710678118654752f));   // exact GELU
}

// ---------------------------------------------------------------------------
// RMSNorm over DMODEL, write f16 (one wave per row, shuffle reduction)
// ---------------------------------------------------------------------------
__global__ __launch_bounds__(256)
void rmsnorm_f16(const float* __restrict__ X, const float* __restrict__ w,
                 _Float16* __restrict__ out)
{
    int lane = threadIdx.x & 31;
    int wave = threadIdx.x >> 5;
    int row  = blockIdx.x * 8 + wave;
    if (row >= MTOT) return;
    const float* xr = X + (size_t)row * DMODEL;
    float s = 0.f;
    float vals[DMODEL / 32];
    #pragma unroll
    for (int j = 0; j < DMODEL / 32; ++j) {
        float v = xr[lane + j * 32];
        vals[j] = v;
        s += v * v;
    }
    #pragma unroll
    for (int off = 16; off >= 1; off >>= 1) s += __shfl_xor(s, off, 32);
    float inv = rsqrtf(s * (1.f / DMODEL) + 1e-5f);
    _Float16* orow = out + (size_t)row * DMODEL;
    #pragma unroll
    for (int j = 0; j < DMODEL / 32; ++j)
        orow[lane + j * 32] = (_Float16)(vals[j] * inv * w[lane + j * 32]);
}

// Final-norm variant: write only 1/rms per row
__global__ __launch_bounds__(256)
void rms_inv(const float* __restrict__ X, float* __restrict__ invr)
{
    int lane = threadIdx.x & 31;
    int wave = threadIdx.x >> 5;
    int row  = blockIdx.x * 8 + wave;
    if (row >= MTOT) return;
    const float* xr = X + (size_t)row * DMODEL;
    float s = 0.f;
    #pragma unroll
    for (int j = 0; j < DMODEL / 32; ++j) { float v = xr[lane + j * 32]; s += v * v; }
    #pragma unroll
    for (int off = 16; off >= 1; off >>= 1) s += __shfl_xor(s, off, 32);
    if (lane == 0) invr[row] = rsqrtf(s * (1.f / DMODEL) + 1e-5f);
}

// ---------------------------------------------------------------------------
// Causal depthwise conv (K=4, left pad 3) + bias + SiLU over u = xz[:, :DINNER]
// writes f32 (scan input) and f16 (x_proj GEMM input)
// ---------------------------------------------------------------------------
__global__ void conv_silu(const float* __restrict__ xz, const float* __restrict__ cw,
                          const float* __restrict__ cb, float* __restrict__ ucf,
                          _Float16* __restrict__ uch)
{
    int i = blockIdx.x * blockDim.x + threadIdx.x;
    if (i >= MTOT * DINNER) return;
    int e = i % DINNER;
    int m = i / DINNER;
    int b = m / SEQL;
    int l = m % SEQL;
    float acc = cb[e];
    #pragma unroll
    for (int k = 0; k < DCONV; ++k) {
        int ls = l - (DCONV - 1) + k;
        if (ls >= 0)
            acc += cw[e * DCONV + k] * xz[((size_t)(b * SEQL + ls)) * (2 * DINNER) + e];
    }
    float s = acc / (1.f + expf(-acc));   // SiLU
    ucf[i] = s;
    uch[i] = (_Float16)s;
}

// Split dt_raw (first DTRANK cols of x_dbl) into an f16 buffer for the dt GEMM
__global__ void split_dtraw(const float* __restrict__ xdbl, _Float16* __restrict__ dtraw)
{
    int i = blockIdx.x * blockDim.x + threadIdx.x;
    if (i >= MTOT * DTRANK) return;
    int r = i % DTRANK;
    int m = i / DTRANK;
    dtraw[i] = (_Float16)xdbl[(size_t)m * (DTRANK + 2 * DSTATE) + r];
}

// ---------------------------------------------------------------------------
// Selective scan: sequential over L, parallel over (b, e). Gating with silu(z)
// fused into the epilogue; writes f16 for the out_proj GEMM.
// grid = (DINNER/256, BATCH), block = 256
// ---------------------------------------------------------------------------
__global__ __launch_bounds__(256)
void selective_scan(const float* __restrict__ dt, const float* __restrict__ ucf,
                    const float* __restrict__ xz,   // z at column offset DINNER
                    const float* __restrict__ xdbl, // Bm at +DTRANK, Cm at +DTRANK+DSTATE
                    const float* __restrict__ A_log, const float* __restrict__ Dp,
                    _Float16* __restrict__ yg)
{
    const int e = blockIdx.x * blockDim.x + threadIdx.x;
    const int b = blockIdx.y;
    if (e >= DINNER) return;

    float Av[DSTATE], hs[DSTATE];
    #pragma unroll
    for (int n = 0; n < DSTATE; ++n) {
        Av[n] = -expf(A_log[(size_t)e * DSTATE + n]);
        hs[n] = 0.f;
    }
    const float Dv = Dp[e];

    for (int t = 0; t < SEQL; ++t) {
        const size_t m  = (size_t)b * SEQL + t;
        const float dtv = dt[m * DINNER + e];
        const float uv  = ucf[m * DINNER + e];
        const float du  = dtv * uv;
        const float* bc = xdbl + m * (DTRANK + 2 * DSTATE) + DTRANK;
        float y = 0.f;
        #pragma unroll
        for (int n = 0; n < DSTATE; ++n) {
            float dA = expf(dtv * Av[n]);
            hs[n] = hs[n] * dA + du * bc[n];
            y += hs[n] * bc[DSTATE + n];
        }
        y += uv * Dv;
        const float z = xz[m * (2 * DINNER) + DINNER + e];
        y *= z / (1.f + expf(-z));        // y * silu(z)
        yg[m * DINNER + e] = (_Float16)y;
    }
}

// ---------------------------------------------------------------------------
// Final: pooled[b,d] = mean_l h[b,l,d]*invrms[b,l]*normf_w[d]; then FC
// ---------------------------------------------------------------------------
__global__ void pool_mean(const float* __restrict__ h, const float* __restrict__ invr,
                          const float* __restrict__ nfw, float* __restrict__ pooled)
{
    int i = blockIdx.x * blockDim.x + threadIdx.x;
    if (i >= BATCH * DMODEL) return;
    int d = i % DMODEL;
    int b = i / DMODEL;
    float s = 0.f;
    for (int l = 0; l < SEQL; ++l) {
        size_t m = (size_t)b * SEQL + l;
        s += h[m * DMODEL + d] * invr[m];
    }
    pooled[i] = s * (1.f / SEQL) * nfw[d];
}

__global__ void fc_head(const float* __restrict__ pooled, const float* __restrict__ fcw,
                        const float* __restrict__ fcb, float* __restrict__ out)
{
    int i = blockIdx.x * blockDim.x + threadIdx.x;
    if (i >= BATCH * NCLS) return;
    int c = i % NCLS;
    int b = i / NCLS;
    float s = fcb[c];
    for (int d = 0; d < DMODEL; ++d)
        s += pooled[b * DMODEL + d] * fcw[c * DMODEL + d];
    out[i] = s;
}

// ---------------------------------------------------------------------------
// Orchestration
// ---------------------------------------------------------------------------
static inline int ceil_div(int a, int b) { return (a + b - 1) / b; }

extern "C" void kernel_launch(void* const* d_in, const int* in_sizes, int n_in,
                              void* d_out, int out_size, void* d_ws, size_t ws_size,
                              hipStream_t stream)
{
    const float* x      = (const float*)d_in[0];
    const float* dw_w   = (const float*)d_in[1];
    const float* pw_w   = (const float*)d_in[2];
    const float* bn_w   = (const float*)d_in[3];
    const float* bn_b   = (const float*)d_in[4];
    const float* bn_m   = (const float*)d_in[5];
    const float* bn_v   = (const float*)d_in[6];
    const float* norm_w = (const float*)d_in[7];
    const float* inpj_w = (const float*)d_in[8];
    const float* conv_w = (const float*)d_in[9];
    const float* conv_b = (const float*)d_in[10];
    const float* xpj_w  = (const float*)d_in[11];
    const float* dtpj_w = (const float*)d_in[12];
    const float* dtpj_b = (const float*)d_in[13];
    const float* A_log  = (const float*)d_in[14];
    const float* Dp     = (const float*)d_in[15];
    const float* outp_w = (const float*)d_in[16];
    const float* normf  = (const float*)d_in[17];
    const float* fc_w   = (const float*)d_in[18];
    const float* fc_b   = (const float*)d_in[19];
    float* out = (float*)d_out;

    // ---- workspace carving (256B aligned) ----
    size_t off = 0;
    auto alloc = [&](size_t bytes) -> char* {
        off = (off + 255) & ~(size_t)255;
        char* p = (char*)d_ws + off;
        off += bytes;
        return p;
    };
    float*     tmp    = (float*)    alloc((size_t)BATCH * CIN_ * SEQL * 4);
    float*     hbuf   = (float*)    alloc((size_t)MTOT * DMODEL * 4);
    _Float16*  xn     = (_Float16*) alloc((size_t)MTOT * DMODEL * 2);
    float*     xz     = (float*)    alloc((size_t)MTOT * 2 * DINNER * 4);
    float*     ucf    = (float*)    alloc((size_t)MTOT * DINNER * 4);
    _Float16*  uch    = (_Float16*) alloc((size_t)MTOT * DINNER * 2);
    float*     xdbl   = (float*)    alloc((size_t)MTOT * (DTRANK + 2 * DSTATE) * 4);
    _Float16*  dtraw  = (_Float16*) alloc((size_t)MTOT * DTRANK * 2);
    float*     dtbuf  = (float*)    alloc((size_t)MTOT * DINNER * 4);
    _Float16*  ygbuf  = (_Float16*) alloc((size_t)MTOT * DINNER * 2);
    float*     invr   = (float*)    alloc((size_t)MTOT * 4);
    float*     pooled = (float*)    alloc((size_t)BATCH * DMODEL * 4);
    _Float16*  w16_in = (_Float16*) alloc((size_t)NLAYER * 2 * DINNER * DMODEL * 2);
    _Float16*  w16_xp = (_Float16*) alloc((size_t)NLAYER * (DTRANK + 2 * DSTATE) * DINNER * 2);
    _Float16*  w16_dt = (_Float16*) alloc((size_t)NLAYER * DINNER * DTRANK * 2);
    _Float16*  w16_op = (_Float16*) alloc((size_t)NLAYER * DMODEL * DINNER * 2);
    (void)ws_size;

    const int TB = 256;

    // ---- convert weights to f16 once per call ----
    {
        int n;
        n = NLAYER * 2 * DINNER * DMODEL;
        cvt_f32_f16<<<ceil_div(n, TB), TB, 0, stream>>>(inpj_w, w16_in, n);
        n = NLAYER * (DTRANK + 2 * DSTATE) * DINNER;
        cvt_f32_f16<<<ceil_div(n, TB), TB, 0, stream>>>(xpj_w, w16_xp, n);
        n = NLAYER * DINNER * DTRANK;
        cvt_f32_f16<<<ceil_div(n, TB), TB, 0, stream>>>(dtpj_w, w16_dt, n);
        n = NLAYER * DMODEL * DINNER;
        cvt_f32_f16<<<ceil_div(n, TB), TB, 0, stream>>>(outp_w, w16_op, n);
    }

    // ---- stem ----
    stem_dw<<<ceil_div(BATCH * CIN_ * SEQL, TB), TB, 0, stream>>>(x, dw_w, tmp);
    stem_pw<<<ceil_div(MTOT * DMODEL, TB), TB, 0, stream>>>(tmp, pw_w, bn_w, bn_b, bn_m, bn_v, hbuf);

    // ---- mamba layers ----
    for (int layer = 0; layer < NLAYER; ++layer) {
        // RMSNorm -> f16
        rmsnorm_f16<<<MTOT / 8, 256, 0, stream>>>(hbuf, norm_w + layer * DMODEL, xn);

        // in_proj: (MTOT,512) x (2048,512)^T -> xz (MTOT,2048)
        mamba_gemm_wmma<<<dim3(MTOT / 128, (2 * DINNER) / 64), 128, 0, stream>>>(
            xn, w16_in + (size_t)layer * 2 * DINNER * DMODEL, xz, nullptr,
            MTOT, 2 * DINNER, DMODEL, 0);

        // causal depthwise conv + SiLU on u half
        conv_silu<<<ceil_div(MTOT * DINNER, TB), TB, 0, stream>>>(
            xz, conv_w + (size_t)layer * DINNER * DCONV, conv_b + layer * DINNER, ucf, uch);

        // x_proj: (MTOT,1024) x (64,1024)^T -> x_dbl (MTOT,64)
        mamba_gemm_wmma<<<dim3(MTOT / 128, (DTRANK + 2 * DSTATE) / 64), 128, 0, stream>>>(
            uch, w16_xp + (size_t)layer * (DTRANK + 2 * DSTATE) * DINNER, xdbl, nullptr,
            MTOT, DTRANK + 2 * DSTATE, DINNER, 0);

        // dt_raw -> f16
        split_dtraw<<<ceil_div(MTOT * DTRANK, TB), TB, 0, stream>>>(xdbl, dtraw);

        // dt = softplus(dt_raw x dtproj^T + b): (MTOT,32) x (1024,32)^T -> (MTOT,1024)
        mamba_gemm_wmma<<<dim3(MTOT / 128, DINNER / 64), 128, 0, stream>>>(
            dtraw, w16_dt + (size_t)layer * DINNER * DTRANK, dtbuf,
            dtpj_b + layer * DINNER, MTOT, DINNER, DTRANK, 2);

        // selective scan + gating -> yg (f16)
        selective_scan<<<dim3(DINNER / 256, BATCH), 256, 0, stream>>>(
            dtbuf, ucf, xz, xdbl,
            A_log + (size_t)layer * DINNER * DSTATE, Dp + layer * DINNER, ygbuf);

        // out_proj: (MTOT,1024) x (512,1024)^T, residual add into h
        mamba_gemm_wmma<<<dim3(MTOT / 128, DMODEL / 64), 128, 0, stream>>>(
            ygbuf, w16_op + (size_t)layer * DMODEL * DINNER, hbuf, nullptr,
            MTOT, DMODEL, DINNER, 3);
    }

    // ---- final norm + mean pool + FC ----
    rms_inv<<<MTOT / 8, 256, 0, stream>>>(hbuf, invr);
    pool_mean<<<ceil_div(BATCH * DMODEL, TB), TB, 0, stream>>>(hbuf, invr, normf, pooled);
    fc_head<<<ceil_div(BATCH * NCLS, TB), TB, 0, stream>>>(pooled, fc_w, fc_b, out);
}
